// Entropy_2911987826791
// MI455X (gfx1250) — matrix operand order repaired and checked by reading, not compiled, not run on hardware
//
#include <hip/hip_runtime.h>

typedef __attribute__((ext_vector_type(16))) _Float16 v16h;
typedef __attribute__((ext_vector_type(8)))  float    v8f;

namespace {
constexpr int   PATCH = 16;
constexpr int   H = 512, W = 512;
constexpr int   HWP = 32;                 // patches per row/col
constexpr float SIGMA = 0.01f;
constexpr float EPS = 1e-40f;
// exp(-0.5*(r/sigma)^2) = exp2(K2 * r * r)
constexpr float K2   = -0.5f / (SIGMA * SIGMA) * 1.4426950408889634f;
constexpr float BIN0 = -1.0f;
constexpr float DBIN = 2.0f / 31.0f;
constexpr float LN2  = 0.6931471805599453f;
}

// Single-instruction 32-lane XOR butterfly via ds_swizzle (group-of-32 mode:
// offset = and_mask(0x1f) | or_mask(0)<<5 | xor_mask<<10). Masks 1/2/4/8 stay
// within each 16-lane half, which is what the C/D bin layout needs.
#define SWZ_XOR(x, mask)                                                      \
  __int_as_float(__builtin_amdgcn_ds_swizzle(__float_as_int(x),               \
                                             (0x1f | ((mask) << 10))))

// One wave handles 16 consecutive patches (a 16x256 gray strip).
// pdf sums are computed as D(16 patches x 16 bins) += A(16x32) x B(32x16)
// with V_WMMA_F32_16X16X32_F16:
//   A[n][k] = (k>>1 == n)          constant patch-selector (2 pixels/patch/chunk)
//   B[k][m] = exp2(K2*(v[p][i]-bin_m)^2), p = k>>1, i = 2c + (k&1)
// Two WMMAs per chunk (bins 0..15 / 16..31), 128 chunks cover 256 pixels.
__global__ __launch_bounds__(64) void entropy_wmma_kernel(const float* __restrict__ in,
                                                          float* __restrict__ out) {
  __shared__ float lds[2][PATCH][256];

  const int wave = threadIdx.x >> 5;
  const int lane = threadIdx.x & 31;
  const int gid  = blockIdx.x * 2 + wave;   // 2048 groups: (b, lh, half)
  const int b    = gid >> 6;
  const int lh   = (gid >> 1) & 31;
  const int half = gid & 1;

  // ---------------- stage: RGB -> gray for a 16 x 256 strip into LDS ----------------
  const float* Rp = in + (size_t)b * 3 * H * W + (size_t)(lh * PATCH) * W + half * 256;
  const float* Gp = Rp + (size_t)H * W;
  const float* Bp = Gp + (size_t)H * W;
  for (int t = lane; t < PATCH * 64; t += 32) {   // 1024 float4 tiles
    const int row = t >> 6;
    const int c4  = t & 63;
    const float4 r4 = *(const float4*)(Rp + row * W + c4 * 4);
    const float4 g4 = *(const float4*)(Gp + row * W + c4 * 4);
    const float4 b4 = *(const float4*)(Bp + row * W + c4 * 4);
    float4 y;
    y.x = 0.2989f * r4.x + 0.587f * g4.x + 0.114f * b4.x;
    y.y = 0.2989f * r4.y + 0.587f * g4.y + 0.114f * b4.y;
    y.z = 0.2989f * r4.z + 0.587f * g4.z + 0.114f * b4.z;
    y.w = 0.2989f * r4.w + 0.587f * g4.w + 0.114f * b4.w;
    *(float4*)&lds[wave][row][c4 * 4] = y;
  }
  __syncthreads();

  // ---------------- compute: KDE histogram via WMMA segmented reduction ----------------
  const int m = lane & 15;        // bin column (N) and A row (M) index for this lane
  const int h = lane >> 4;        // lane half (selects K sub-ranges per ISA layouts)
  const float binLo = BIN0 + DBIN * (float)m;
  const float binHi = BIN0 + DBIN * (float)(m + 16);

  // A-matrix (16x32 f16): lane L holds row M=L&15; element e maps to
  // K = e + 16*h  (contiguous order, matching B).
  v16h A;
#pragma unroll
  for (int e = 0; e < 16; ++e) {
    const int k = e + 16 * h;
    A[e] = ((k >> 1) == m) ? (_Float16)1.0f : (_Float16)0.0f;
  }

  v8f accLo = {};
  v8f accHi = {};
  const float* V = &lds[wave][0][0];

  for (int c = 0; c < 128; ++c) {
    const int ir = c >> 3;            // row within patch
    const int ic = (c & 7) << 1;      // even column within patch
    // B-matrix (32x16 f16): lane L holds column N=L&15; element e maps to
    // K = e + 16*h. Elements (2j, 2j+1) share patch p = j + 8h, parities 0/1.
    v16h Blo, Bhi;
#pragma unroll
    for (int j = 0; j < 8; ++j) {
      const int p = j + 8 * h;
      const float2 pv = *(const float2*)(V + ir * 256 + p * 16 + ic);
      const float r0 = pv.x - binLo;
      const float r1 = pv.y - binLo;
      const float s0 = pv.x - binHi;
      const float s1 = pv.y - binHi;
      Blo[2 * j]     = (_Float16)__builtin_amdgcn_exp2f(K2 * r0 * r0);
      Blo[2 * j + 1] = (_Float16)__builtin_amdgcn_exp2f(K2 * r1 * r1);
      Bhi[2 * j]     = (_Float16)__builtin_amdgcn_exp2f(K2 * s0 * s0);
      Bhi[2 * j + 1] = (_Float16)__builtin_amdgcn_exp2f(K2 * s1 * s1);
    }
    accLo = __builtin_amdgcn_wmma_f32_16x16x32_f16(false, A, false, Blo, (short)0, accLo,
                                                   false, false);
    accHi = __builtin_amdgcn_wmma_f32_16x16x32_f16(false, A, false, Bhi, (short)0, accHi,
                                                   false, false);
  }

  // ---------------- normalize + entropy ----------------
  // D layout: VGPR r, lane L -> patch M = r + 8*(L>>4), bin N = (L&15) (+16 for accHi).
  const int outBase = b * (HWP * HWP) + lh * HWP + half * 16 + 8 * h;
#pragma unroll
  for (int r = 0; r < 8; ++r) {
    const float pLo = accLo[r] * (1.0f / 256.0f);   // pdf = mean over 256 pixels
    const float pHi = accHi[r] * (1.0f / 256.0f);
    float s = pLo + pHi;                            // sum over the 32 bins (16 lanes x 2)
    s += SWZ_XOR(s, 1);
    s += SWZ_XOR(s, 2);
    s += SWZ_XOR(s, 4);
    s += SWZ_XOR(s, 8);
    const float inv = __builtin_amdgcn_rcpf(s + EPS);   // v_rcp_f32, ~1 ulp
    const float pnLo = pLo * inv + EPS;
    const float pnHi = pHi * inv + EPS;
    // natural log via v_log_f32 (log2) * ln2
    float t = pnLo * __builtin_amdgcn_logf(pnLo) + pnHi * __builtin_amdgcn_logf(pnHi);
    t += SWZ_XOR(t, 1);
    t += SWZ_XOR(t, 2);
    t += SWZ_XOR(t, 4);
    t += SWZ_XOR(t, 8);
    if (m == 0) out[outBase + r] = -LN2 * t;
  }
}

extern "C" void kernel_launch(void* const* d_in, const int* in_sizes, int n_in,
                              void* d_out, int out_size, void* d_ws, size_t ws_size,
                              hipStream_t stream) {
  const float* in = (const float*)d_in[0];
  float* out = (float*)d_out;
  // 2048 wave-groups, 2 waves per 64-thread block -> 1024 blocks
  entropy_wmma_kernel<<<dim3(1024), dim3(64), 0, stream>>>(in, out);
}
